// Block_68805376082476
// MI455X (gfx1250) — compile-verified, matrix operand order checked by reading, not run on hardware
//
#include <hip/hip_runtime.h>
#include <math.h>

// ---------------------------------------------------------------------------
// Problem constants (GPT block: B=4, T=2048, d=768, H=12, hd=64)
// ---------------------------------------------------------------------------
#define NB 4
#define TT 2048
#define DD 768
#define NH 12
#define HD 64
#define MM (NB * TT)          // 8192 token rows

typedef __attribute__((ext_vector_type(16))) __bf16 v16bf;
typedef __attribute__((ext_vector_type(8)))  __bf16 v8bf;
typedef __attribute__((ext_vector_type(8)))  float  v8f;

union FragBF {
    v16bf  v;
    v8bf   h[2];
    __bf16 e[16];
};

__device__ inline v8f zero8() {
    v8f z = {0.f, 0.f, 0.f, 0.f, 0.f, 0.f, 0.f, 0.f};
    return z;
}

// 16-bit A/B fragment layout (CDNA5 ISA 7.12.2):
//   lane = half*16 + row;  element j: K = j + 8*half          (j < 8)
//                                     K = j + 8 + 8*half      (j >= 8)
// -> two contiguous 16B loads per lane at K offsets 8*half and 16+8*half.
__device__ inline v16bf load_frag_ptr(const __bf16* p, int half) {
    FragBF f;
    f.h[0] = *(const v8bf*)(p + 8 * half);
    f.h[1] = *(const v8bf*)(p + 16 + 8 * half);
    return f.v;
}

__device__ inline v16bf load_frag_rowmajor(const __bf16* __restrict__ base,
                                           int row, int ld, int k0, int half) {
    return load_frag_ptr(base + (size_t)row * ld + k0, half);
}

__device__ inline v8f wmma_bf16(v16bf a, v16bf b, v8f c) {
    // (neg_a, A, neg_b, B, c_mod, C, reuse_a, reuse_b)
    return __builtin_amdgcn_wmma_f32_16x16x32_bf16(false, a, false, b,
                                                   (short)0, c, false, false);
}

// ---------------------------------------------------------------------------
// Weight convert+transpose: W[K][N] fp32 -> WT[N][K] bf16
// ---------------------------------------------------------------------------
__global__ void convert_wT_kernel(const float* __restrict__ W,
                                  __bf16* __restrict__ WT,
                                  int K, int N) {
    size_t idx = (size_t)blockIdx.x * blockDim.x + threadIdx.x;
    size_t total = (size_t)K * N;
    if (idx >= total) return;
    int k = (int)(idx / N);
    int n = (int)(idx % N);
    WT[(size_t)n * K + k] = (__bf16)W[idx];
}

// ---------------------------------------------------------------------------
// LayerNorm (row of 768 fp32) -> bf16
// ---------------------------------------------------------------------------
__global__ __launch_bounds__(256)
void layernorm_bf16_kernel(const float* __restrict__ x,
                           const float* __restrict__ g,
                           const float* __restrict__ b,
                           __bf16* __restrict__ out, int C) {
    const int row = blockIdx.x;
    const float* xr = x + (size_t)row * C;
    __shared__ float ssum[256];
    __shared__ float ssq[256];
    float s = 0.f, q = 0.f;
    for (int c = threadIdx.x; c < C; c += blockDim.x) {
        float v = xr[c];
        s += v;
        q += v * v;
    }
    ssum[threadIdx.x] = s;
    ssq[threadIdx.x]  = q;
    __syncthreads();
    for (int off = 128; off > 0; off >>= 1) {
        if ((int)threadIdx.x < off) {
            ssum[threadIdx.x] += ssum[threadIdx.x + off];
            ssq[threadIdx.x]  += ssq[threadIdx.x + off];
        }
        __syncthreads();
    }
    const float mean = ssum[0] / (float)C;
    const float var  = ssq[0] / (float)C - mean * mean;
    const float inv  = rsqrtf(var + 1e-5f);
    for (int c = threadIdx.x; c < C; c += blockDim.x) {
        float v = (xr[c] - mean) * inv * g[c] + b[c];
        out[(size_t)row * C + c] = (__bf16)v;
    }
}

// ---------------------------------------------------------------------------
// WMMA GEMM with async global->LDS double-buffered staging.
// Block = 256 threads (8 waves). Block tile 128(M) x 128(N), K-slab 32.
// Each wave computes a 32x64 subtile: 8 WMMAs per K-slab from LDS frags.
// Async slab: 4 x global_load_async_to_lds_b128 per thread (A:2, B:2).
// ---------------------------------------------------------------------------
#define BM 128
#define BN 128
#define LDSROW 40   // padded LDS row stride in elements (80B: 16B-aligned, bank-spread)

enum { EPI_BF16 = 0, EPI_F32_RES = 1, EPI_GELU_BF16 = 2 };

template <int EPI>
__global__ __launch_bounds__(256)
void gemm_wmma_kernel(const __bf16* __restrict__ A,
                      const __bf16* __restrict__ WT,
                      const float*  __restrict__ bias,
                      const float*  __restrict__ res,
                      float*        __restrict__ outF,
                      __bf16*       __restrict__ outH,
                      int M, int N, int K) {
    __shared__ __bf16 As[2][BM * LDSROW];
    __shared__ __bf16 Bs[2][BN * LDSROW];

    const int tid  = threadIdx.x;
    const int lane = tid & 31;
    const int wv   = tid >> 5;          // wave 0..7
    const int n15  = lane & 15;
    const int half = lane >> 4;
    const int wm   = (wv & 3) * 32;     // wave M offset inside block tile
    const int wn   = (wv >> 2) * 64;    // wave N offset inside block tile
    const int n0   = blockIdx.x * BN;
    const int m0   = blockIdx.y * BM;

    // Issue one 128x32 bf16 slab (8KB) as 2 async b128 chunks per thread.
    // chunk id c = r*256 + tid : row = c>>2 (0..127), kchunk = c&3 (8 elems).
    auto issue_tile = [&](const __bf16* __restrict__ gbase, int grow0, int k0,
                          __bf16* lds) {
#pragma unroll
        for (int r = 0; r < 2; ++r) {
            const int row = r * 64 + (tid >> 2);
            const int kc  = tid & 3;
            const unsigned goff =
                (unsigned)(((size_t)(grow0 + row) * K + (size_t)k0 + kc * 8) * 2);
            const unsigned loff =
                (unsigned)(uintptr_t)(lds + row * LDSROW + kc * 8);
            // GVS mode: vdst = LDS byte address, vaddr = 32b offset, saddr = base
            asm volatile("global_load_async_to_lds_b128 %0, %1, %2"
                         :: "v"(loff), "v"(goff), "s"(gbase)
                         : "memory");
        }
    };

    v8f acc[2][4];
#pragma unroll
    for (int i = 0; i < 2; i++)
#pragma unroll
        for (int j = 0; j < 4; j++) acc[i][j] = zero8();

    const int nsteps = K / 32;

    issue_tile(A,  m0, 0, &As[0][0]);
    issue_tile(WT, n0, 0, &Bs[0][0]);

    for (int step = 0; step < nsteps; ++step) {
        const int c = step & 1;
        if (step + 1 < nsteps) {
            // prefetch next slab into the other buffer (last read in the previous step,
            // protected by the end-of-step barrier)
            issue_tile(A,  m0, (step + 1) * 32, &As[1 - c][0]);
            issue_tile(WT, n0, (step + 1) * 32, &Bs[1 - c][0]);
            // in-order completion: <=4 outstanding => current slab landed
            asm volatile("s_wait_asynccnt 0x4" ::: "memory");
        } else {
            asm volatile("s_wait_asynccnt 0x0" ::: "memory");
        }
        __syncthreads();   // all waves' chunks of slab `c` visible

        v16bf a0 = load_frag_ptr(&As[c][(wm + n15) * LDSROW],      half);
        v16bf a1 = load_frag_ptr(&As[c][(wm + 16 + n15) * LDSROW], half);
        v16bf b0 = load_frag_ptr(&Bs[c][(wn + n15) * LDSROW],      half);
        v16bf b1 = load_frag_ptr(&Bs[c][(wn + 16 + n15) * LDSROW], half);
        v16bf b2 = load_frag_ptr(&Bs[c][(wn + 32 + n15) * LDSROW], half);
        v16bf b3 = load_frag_ptr(&Bs[c][(wn + 48 + n15) * LDSROW], half);

        acc[0][0] = wmma_bf16(a0, b0, acc[0][0]);
        acc[0][1] = wmma_bf16(a0, b1, acc[0][1]);
        acc[0][2] = wmma_bf16(a0, b2, acc[0][2]);
        acc[0][3] = wmma_bf16(a0, b3, acc[0][3]);
        acc[1][0] = wmma_bf16(a1, b0, acc[1][0]);
        acc[1][1] = wmma_bf16(a1, b1, acc[1][1]);
        acc[1][2] = wmma_bf16(a1, b2, acc[1][2]);
        acc[1][3] = wmma_bf16(a1, b3, acc[1][3]);

        __syncthreads();   // everyone done reading slab `c` before it is refilled
    }

    // C/D layout: n = lane&15, m = vgpr + 8*half
#pragma unroll
    for (int i = 0; i < 2; i++)
#pragma unroll
        for (int j = 0; j < 4; j++)
#pragma unroll
            for (int e = 0; e < 8; e++) {
                const int m = m0 + wm + i * 16 + e + 8 * half;
                const int n = n0 + wn + j * 16 + n15;
                float v = acc[i][j][e] + bias[n];
                if (EPI == EPI_F32_RES) {
                    v += res[(size_t)m * N + n];
                    outF[(size_t)m * N + n] = v;
                } else if (EPI == EPI_GELU_BF16) {
                    v = 0.5f * v * (1.0f + erff(v * 0.70710678118654752f));
                    outH[(size_t)m * N + n] = (__bf16)v;
                } else {
                    outH[(size_t)m * N + n] = (__bf16)v;
                }
            }
}

// ---------------------------------------------------------------------------
// V transpose: qkv[B*T][2304] (v at col 1536 + h*64 + d) -> Vt[b][h][d][T]
// ---------------------------------------------------------------------------
__global__ void transpose_v_kernel(const __bf16* __restrict__ qkv,
                                   __bf16* __restrict__ Vt) {
    size_t idx = (size_t)blockIdx.x * blockDim.x + threadIdx.x;
    const size_t total = (size_t)MM * DD;
    if (idx >= total) return;
    const int tok = (int)(idx / DD);
    const int c   = (int)(idx % DD);   // c = h*64 + d
    const int b   = tok / TT;
    const int t   = tok % TT;
    const int h   = c / HD;
    const int d   = c % HD;
    Vt[((size_t)(b * NH + h) * HD + d) * TT + t] =
        qkv[(size_t)tok * (3 * DD) + 2 * DD + c];
}

// ---------------------------------------------------------------------------
// Flash attention: per (b, h, 16-query tile) wave; 32-key tiles; causal.
// ---------------------------------------------------------------------------
__global__ __launch_bounds__(32)
void attn_flash_kernel(const __bf16* __restrict__ qkv,
                       const __bf16* __restrict__ Vt,
                       __bf16* __restrict__ y) {
    const int lane = threadIdx.x;
    const int n15  = lane & 15;
    const int half = lane >> 4;
    const int q0   = blockIdx.x * 16;
    const int h    = blockIdx.y;
    const int b    = blockIdx.z;
    const int ldq  = 3 * DD;

    const __bf16* qbase = qkv + (size_t)b * TT * ldq + h * HD;
    const __bf16* kbase = qbase + DD;
    const __bf16* vtb   = Vt + (size_t)(b * NH + h) * HD * TT;

    const v16bf qa0 = load_frag_rowmajor(qbase, q0 + n15, ldq, 0,  half);
    const v16bf qa1 = load_frag_rowmajor(qbase, q0 + n15, ldq, 32, half);

    v8f o[4];
#pragma unroll
    for (int t = 0; t < 4; t++) o[t] = zero8();
    float mrow[8], lrow[8];
#pragma unroll
    for (int i = 0; i < 8; i++) { mrow[i] = -3.0e38f; lrow[i] = 0.0f; }

    __shared__ __bf16 Pl[16 * 32];

    for (int kt = 0; kt <= q0 + 15; kt += 32) {
        // S = Q K^T (16x32 scores) : 4 WMMAs
        v16bf kb00 = load_frag_rowmajor(kbase, kt + n15,      ldq, 0,  half);
        v16bf kb01 = load_frag_rowmajor(kbase, kt + n15,      ldq, 32, half);
        v16bf kb10 = load_frag_rowmajor(kbase, kt + 16 + n15, ldq, 0,  half);
        v16bf kb11 = load_frag_rowmajor(kbase, kt + 16 + n15, ldq, 32, half);
        v8f s0 = zero8(), s1 = zero8();
        s0 = wmma_bf16(qa0, kb00, s0);
        s0 = wmma_bf16(qa1, kb01, s0);
        s1 = wmma_bf16(qa0, kb10, s1);
        s1 = wmma_bf16(qa1, kb11, s1);

        const bool need_mask = (kt + 31 > q0);  // wave-uniform

        // causal mask + online softmax (row stats via lane-group butterflies)
#pragma unroll
        for (int i = 0; i < 8; i++) {
            const int m = q0 + i + 8 * half;
            float v0 = s0[i] * 0.125f;               // 1/sqrt(64)
            float v1 = s1[i] * 0.125f;
            if (need_mask) {
                if (kt + n15 > m)      v0 = -3.0e38f;
                if (kt + 16 + n15 > m) v1 = -3.0e38f;
            }

            float t = fmaxf(v0, v1);
            t = fmaxf(t, __shfl_xor(t, 1, 32));
            t = fmaxf(t, __shfl_xor(t, 2, 32));
            t = fmaxf(t, __shfl_xor(t, 4, 32));
            t = fmaxf(t, __shfl_xor(t, 8, 32));
            const float nm = fmaxf(mrow[i], t);

            const float p0 = __expf(v0 - nm);
            const float p1 = __expf(v1 - nm);
            float rs = p0 + p1;
            rs += __shfl_xor(rs, 1, 32);
            rs += __shfl_xor(rs, 2, 32);
            rs += __shfl_xor(rs, 4, 32);
            rs += __shfl_xor(rs, 8, 32);

            const float corr = __expf(mrow[i] - nm);
            lrow[i] = lrow[i] * corr + rs;
            mrow[i] = nm;
            o[0][i] *= corr;
            o[1][i] *= corr;
            o[2][i] *= corr;
            o[3][i] *= corr;

            // stash P in LDS (C layout -> A layout transpose through LDS)
            Pl[(i + 8 * half) * 32 + n15]      = (__bf16)p0;
            Pl[(i + 8 * half) * 32 + 16 + n15] = (__bf16)p1;
        }
        __syncthreads();
        asm volatile("s_wait_dscnt 0" ::: "memory");

        FragBF pf;
        pf.h[0] = *(const v8bf*)(&Pl[n15 * 32 + 8 * half]);
        pf.h[1] = *(const v8bf*)(&Pl[n15 * 32 + 16 + 8 * half]);

        // O += P V : 4 WMMAs against pre-transposed V^T rows (contiguous)
#pragma unroll
        for (int t = 0; t < 4; t++) {
            v16bf vb = load_frag_rowmajor(vtb, t * 16 + n15, TT, kt, half);
            o[t] = wmma_bf16(pf.v, vb, o[t]);
        }
        __syncthreads();
    }

#pragma unroll
    for (int t = 0; t < 4; t++)
#pragma unroll
        for (int i = 0; i < 8; i++) {
            const int m = q0 + i + 8 * half;
            const float v = o[t][i] / lrow[i];
            y[((size_t)b * TT + m) * DD + h * HD + t * 16 + n15] = (__bf16)v;
        }
}

// ---------------------------------------------------------------------------
// Host launcher
// ---------------------------------------------------------------------------
extern "C" void kernel_launch(void* const* d_in, const int* in_sizes, int n_in,
                              void* d_out, int out_size, void* d_ws, size_t ws_size,
                              hipStream_t stream) {
    const float* x       = (const float*)d_in[0];
    const float* ln1_g   = (const float*)d_in[1];
    const float* ln1_b   = (const float*)d_in[2];
    const float* w_attn  = (const float*)d_in[3];
    const float* b_attn  = (const float*)d_in[4];
    const float* w_aproj = (const float*)d_in[5];
    const float* b_aproj = (const float*)d_in[6];
    const float* ln2_g   = (const float*)d_in[7];
    const float* ln2_b   = (const float*)d_in[8];
    const float* w_fc    = (const float*)d_in[9];
    const float* b_fc    = (const float*)d_in[10];
    const float* w_mproj = (const float*)d_in[11];
    const float* b_mproj = (const float*)d_in[12];
    float* out = (float*)d_out;

    char* ws = (char*)d_ws;
    size_t off = 0;
    auto alloc = [&](size_t bytes) -> void* {
        void* p = ws + off;
        off += (bytes + 255) & ~(size_t)255;
        return p;
    };
    __bf16* wt_attn  = (__bf16*)alloc((size_t)DD * 3 * DD * 2);
    __bf16* wt_aproj = (__bf16*)alloc((size_t)DD * DD * 2);
    __bf16* wt_fc    = (__bf16*)alloc((size_t)DD * 4 * DD * 2);
    __bf16* wt_mproj = (__bf16*)alloc((size_t)4 * DD * DD * 2);
    __bf16* xn       = (__bf16*)alloc((size_t)MM * DD * 2);      // ln1 out; reused as attn-out y
    __bf16* big      = (__bf16*)alloc((size_t)MM * 4 * DD * 2);  // qkv (3d wide); reused as gelu out (4d wide)
    __bf16* vt_h     = (__bf16*)alloc((size_t)MM * DD * 2);      // V^T; reused as ln2 out
    float*  x1       = (float*) alloc((size_t)MM * DD * 4);      // residual after attention
    (void)ws_size; (void)in_sizes; (void)n_in; (void)out_size;

    // 1) weights -> bf16 transposed
    {
        size_t e1 = (size_t)DD * 3 * DD;
        convert_wT_kernel<<<dim3((unsigned)((e1 + 255) / 256)), 256, 0, stream>>>(w_attn, wt_attn, DD, 3 * DD);
        size_t e2 = (size_t)DD * DD;
        convert_wT_kernel<<<dim3((unsigned)((e2 + 255) / 256)), 256, 0, stream>>>(w_aproj, wt_aproj, DD, DD);
        size_t e3 = (size_t)DD * 4 * DD;
        convert_wT_kernel<<<dim3((unsigned)((e3 + 255) / 256)), 256, 0, stream>>>(w_fc, wt_fc, DD, 4 * DD);
        size_t e4 = (size_t)4 * DD * DD;
        convert_wT_kernel<<<dim3((unsigned)((e4 + 255) / 256)), 256, 0, stream>>>(w_mproj, wt_mproj, 4 * DD, DD);
    }

    // 2) LN1
    layernorm_bf16_kernel<<<MM, 256, 0, stream>>>(x, ln1_g, ln1_b, xn, DD);

    // 3) qkv = LN1(x) @ w_attn + b_attn  -> bf16
    gemm_wmma_kernel<EPI_BF16><<<dim3(3 * DD / BN, MM / BM), 256, 0, stream>>>(
        xn, wt_attn, b_attn, nullptr, nullptr, big, MM, 3 * DD, DD);

    // 4) V^T
    {
        size_t ev = (size_t)MM * DD;
        transpose_v_kernel<<<dim3((unsigned)((ev + 255) / 256)), 256, 0, stream>>>(big, vt_h);
    }

    // 5) flash attention -> y (reuses xn)
    attn_flash_kernel<<<dim3(TT / 16, NH, NB), 32, 0, stream>>>(big, vt_h, xn);

    // 6) x1 = x + y @ w_aproj + b_aproj  (fp32)
    gemm_wmma_kernel<EPI_F32_RES><<<dim3(DD / BN, MM / BM), 256, 0, stream>>>(
        xn, wt_aproj, b_aproj, x, x1, nullptr, MM, DD, DD);

    // 7) LN2 -> h (reuses vt_h)
    layernorm_bf16_kernel<<<MM, 256, 0, stream>>>(x1, ln2_g, ln2_b, vt_h, DD);

    // 8) g = gelu(h @ w_fc + b_fc) -> bf16 (reuses big)
    gemm_wmma_kernel<EPI_GELU_BF16><<<dim3(4 * DD / BN, MM / BM), 256, 0, stream>>>(
        vt_h, wt_fc, b_fc, nullptr, nullptr, big, MM, 4 * DD, DD);

    // 9) out = x1 + g @ w_mproj + b_mproj  (fp32)
    gemm_wmma_kernel<EPI_F32_RES><<<dim3(DD / BN, MM / BM), 256, 0, stream>>>(
        big, wt_mproj, b_mproj, x1, out, nullptr, MM, DD, 4 * DD);
}
